// PagedMultiHeadAttention_43516608643312
// MI455X (gfx1250) — matrix operand order repaired and checked by reading, not compiled, not run on hardware
//
#include <hip/hip_runtime.h>

// ---------------------------------------------------------------------------
// Fused: qkv = q @ Wqkv^T ; causal GQA flash-attention ; out = ctx @ Wdense^T
// B=2, S=2048, EMB=4096, NHEADS=32, KVHEADS=8, HEAD_D=128, G=4
// Matmuls via v_wmma_f32_16x16x32_bf16; GEMMs use async global->LDS staging
// (ASYNCcnt) with double buffering.
// ---------------------------------------------------------------------------

typedef __bf16 bf16_t;
typedef __attribute__((ext_vector_type(16))) __bf16 v16bf;
typedef __attribute__((ext_vector_type(8)))  float  v8f;

#define B_DIM 2
#define S_DIM 2048
#define EMB 4096
#define NHEADS 32
#define KVHEADS 8
#define HEAD_D 128
#define GQA_G 4
#define QKV_N 6144           // 4096 Q + 1024 K + 1024 V
#define M_ROWS (B_DIM * S_DIM)
#define SOFTMAX_SCALE 0.08838834764831845f  // 1/sqrt(128)

// Async global->LDS copy (CDNA5 GLOBAL_LOAD_ASYNC_TO_LDS_B128, ASYNCcnt).
#if defined(__HIP_DEVICE_COMPILE__) && __has_builtin(__builtin_amdgcn_global_load_async_to_lds_b128)
#define HAVE_ASYNC_LDS 1
#else
#define HAVE_ASYNC_LDS 0
#endif

typedef int v4i_cpy __attribute__((vector_size(16)));
typedef __attribute__((address_space(1))) v4i_cpy* gbl_v4p;
typedef __attribute__((address_space(3))) v4i_cpy* lds_v4p;

static __device__ __forceinline__ void copy16_g2l(bf16_t* lds_dst,
                                                  const bf16_t* g_src) {
#if HAVE_ASYNC_LDS
  __builtin_amdgcn_global_load_async_to_lds_b128((gbl_v4p)g_src,
                                                 (lds_v4p)lds_dst, 0, 0);
#else
  *(uint4*)lds_dst = *(const uint4*)g_src;
#endif
}
static __device__ __forceinline__ void wait_async_le4() {
#if HAVE_ASYNC_LDS
  asm volatile("s_wait_asynccnt 0x4" ::: "memory");
#endif
}
static __device__ __forceinline__ void wait_async_le0() {
#if HAVE_ASYNC_LDS
  asm volatile("s_wait_asynccnt 0x0" ::: "memory");
#endif
}

static __device__ __forceinline__ v8f wmma_bf16(v16bf a, v16bf b, v8f c) {
  return __builtin_amdgcn_wmma_f32_16x16x32_bf16(
      /*neg_a=*/false, a, /*neg_b=*/false, b,
      /*c_mod=*/(short)0, c, /*reuse_a=*/false, /*reuse_b=*/false);
}

// A-matrix 16x32 bf16 fragment from row-major [rows, ld] memory.
// Lane layout (ISA 7.12.2): lane m=lane&15 holds row m; VGPR0..3 = K 8h..8h+7,
// VGPR4..7 = K 16+8h..16+8h+7 where h = lane>>4.  -> two contiguous 16B loads.
static __device__ __forceinline__ v16bf load_a_frag(const bf16_t* base, int ld,
                                                    int row0, int k0, int lane) {
  int m = lane & 15, h = lane >> 4;
  const bf16_t* p = base + (size_t)(row0 + m) * (size_t)ld + k0 + 8 * h;
  union { v16bf v; uint4 u[2]; } f;
  f.u[0] = *(const uint4*)p;
  f.u[1] = *(const uint4*)(p + 16);
  return f.v;
}

// B-matrix 32x16 bf16 fragment where memory holds B^T row-major: src[n][k].
// Lane n=lane&15 holds column n; K 8h..8h+7 then K 16+8h..16+8h+7.
static __device__ __forceinline__ v16bf load_b_frag(const bf16_t* base, int ld,
                                                    int n0, int k0, int lane) {
  int n = lane & 15, h = lane >> 4;
  const bf16_t* p = base + (size_t)(n0 + n) * (size_t)ld + k0 + 8 * h;
  union { v16bf v; uint4 u[2]; } f;
  f.u[0] = *(const uint4*)p;
  f.u[1] = *(const uint4*)(p + 16);
  return f.v;
}

// ---------------------------------------------------------------------------
// fp32 -> bf16 conversion (vectorized 4-wide)
// ---------------------------------------------------------------------------
__global__ void cvt_f32_bf16(const float4* __restrict__ in,
                             bf16_t* __restrict__ out, int n4) {
  int i = blockIdx.x * blockDim.x + threadIdx.x;
  if (i < n4) {
    float4 f = in[i];
    union { bf16_t h[4]; uint2 u; } r;
    r.h[0] = (bf16_t)f.x; r.h[1] = (bf16_t)f.y;
    r.h[2] = (bf16_t)f.z; r.h[3] = (bf16_t)f.w;
    *(uint2*)(out + (size_t)i * 4) = r.u;
  }
}

// ---------------------------------------------------------------------------
// GEMM: Out[M,N] = X[M,K] @ W[N,K]^T  (torch nn.Linear layout)
// Block = 256 threads (8 waves), 128x128 tile; wave = 64x32 = 4x2 WMMA tiles.
// K panels (128x32 A and 128x32 B, bf16) staged in LDS via async copies,
// double-buffered; rows padded to 40 halfs (20-bank stride, conflict-free).
// First `scale_ncols` output columns scaled by `scale` (folds softmax scale
// into Q at the QKV store). STORE_BF16 selects bf16 vs fp32 output.
// ---------------------------------------------------------------------------
#define GPAD 40   // 32-half panel row padded to 40 halfs (80B)

template <bool STORE_BF16>
__global__ __launch_bounds__(256)
void gemm_xwt(const bf16_t* __restrict__ X, const bf16_t* __restrict__ W,
              void* __restrict__ Out, int M, int N, int K,
              int scale_ncols, float scale) {
  __shared__ __align__(16) bf16_t Ash[2][128 * GPAD];
  __shared__ __align__(16) bf16_t Bsh[2][128 * GPAD];

  const int tid = threadIdx.x;
  const int wave = tid >> 5, lane = tid & 31;
  const int rowblk = blockIdx.x * 128;
  const int colblk = blockIdx.y * 128;
  const int wrow = (wave >> 2) * 64;   // wave 64-row sub-tile within block
  const int wcol = (wave & 3) * 32;    // wave 32-col sub-tile within block

  // Stage one 128x32 K-panel of A and B into LDS buffer `buf`.
  // 512 16B chunks per matrix; 256 threads x 2 chunks, thread-contiguous.
  auto stage = [&](int s, int buf) {
    const int k0 = s * 32;
#pragma unroll
    for (int r = 0; r < 2; ++r) {
      int c = tid + 256 * r;            // 0..511
      int row = c >> 2, col = (c & 3) * 8;
      copy16_g2l(&Ash[buf][row * GPAD + col],
                 X + (size_t)(rowblk + row) * K + k0 + col);
      copy16_g2l(&Bsh[buf][row * GPAD + col],
                 W + (size_t)(colblk + row) * K + k0 + col);
    }
  };

  v8f acc[4][2] = {};
  const int nstages = K / 32;
  stage(0, 0);
  for (int s = 0; s < nstages; ++s) {
    const int buf = s & 1;
    if (s + 1 < nstages) {
      stage(s + 1, buf ^ 1);   // prefetch next panel (async)
      wait_async_le4();        // this wave's stage-s loads retired (in order)
    } else {
      wait_async_le0();
    }
    __syncthreads();           // all waves' stage-s data visible in LDS

    v16bf a[4], b[2];
#pragma unroll
    for (int i = 0; i < 4; ++i)
      a[i] = load_a_frag(Ash[buf], GPAD, wrow + 16 * i, 0, lane);
#pragma unroll
    for (int j = 0; j < 2; ++j)
      b[j] = load_b_frag(Bsh[buf], GPAD, wcol + 16 * j, 0, lane);
#pragma unroll
    for (int i = 0; i < 4; ++i)
#pragma unroll
      for (int j = 0; j < 2; ++j)
        acc[i][j] = wmma_bf16(a[i], b[j], acc[i][j]);

    __syncthreads();           // panel reads done before it is overwritten
  }

  // C layout: VGPR v holds row (v + 8*(lane>>4)), column lane&15.
  int h = lane >> 4, nn = lane & 15;
#pragma unroll
  for (int i = 0; i < 4; ++i)
#pragma unroll
    for (int j = 0; j < 2; ++j) {
      int col = colblk + wcol + 16 * j + nn;
      float sc = (col < scale_ncols) ? scale : 1.0f;
#pragma unroll
      for (int v = 0; v < 8; ++v) {
        int row = rowblk + wrow + 16 * i + v + 8 * h;
        float val = acc[i][j][v] * sc;
        if constexpr (STORE_BF16)
          ((bf16_t*)Out)[(size_t)row * N + col] = (bf16_t)val;
        else
          ((float*)Out)[(size_t)row * N + col] = val;
      }
    }
}

// ---------------------------------------------------------------------------
// Flash attention, causal, GQA (4 query heads share each kv head).
// Grid: B * KVHEADS * (S/32). Block = 256 threads = 8 waves:
//   wave w -> query head g = w>>1 (within kv group), row-subtile rs = w&1.
// Each wave owns 16 query rows and streams key blocks of 32.
// ---------------------------------------------------------------------------
#define KPAD 136   // 32 x 128 K-block rows padded (+16B) -> conflict-free b128 reads
#define VPAD 40    // 128 x 32 transposed V rows padded
#define PPAD 40    // per-wave 16 x 32 P staging rows padded

__global__ __launch_bounds__(256)
void attn_kernel(const bf16_t* __restrict__ qkv, bf16_t* __restrict__ ctx) {
  __shared__ __align__(16) bf16_t Ksh[32 * KPAD];       // [key][d]
  __shared__ __align__(16) bf16_t Vsh[HEAD_D * VPAD];   // [d][key]  (transposed)
  __shared__ __align__(16) bf16_t Psh[8][16 * PPAD];    // per-wave P staging

  const int mt = blockIdx.x % (S_DIM / 32);
  const int kv = (blockIdx.x / (S_DIM / 32)) % KVHEADS;
  const int b  = blockIdx.x / (S_DIM / 32) / KVHEADS;

  const int tid = threadIdx.x, wave = tid >> 5, lane = tid & 31;
  const int g = wave >> 1, rs = wave & 1;
  const int q0 = mt * 32;                 // workgroup query base
  const int qbase = q0 + rs * 16;         // this wave's 16 query rows
  const size_t rowbase = (size_t)b * S_DIM;
  const int qcol = (kv * GQA_G + g) * HEAD_D;
  const int kcol = EMB + kv * HEAD_D;
  const int vcol = EMB + KVHEADS * HEAD_D + kv * HEAD_D;

  // Preload Q fragments (softmax scale already folded into Q at QKV store).
  v16bf aq[4];
#pragma unroll
  for (int c = 0; c < 4; ++c)
    aq[c] = load_a_frag(qkv + qcol, QKV_N, (int)rowbase + qbase, c * 32, lane);

  float m_run[8], l_run[8];
  v8f acc[8] = {};
#pragma unroll
  for (int v = 0; v < 8; ++v) { m_run[v] = -1e30f; l_run[v] = 0.0f; }

  const int h = lane >> 4, nn = lane & 15;
  const int jmax = mt;  // inclusive: key blocks 0..mt cover keys <= q0+31

  for (int j = 0; j <= jmax; ++j) {
    __syncthreads();  // previous iteration's Vsh reads complete
    // Cooperative K/V block load: keys j*32 .. j*32+31, 128 d each.
    for (int c = tid; c < 512; c += 256) {
      int kr = c >> 4;
      int dc = (c & 15) * 8;
      const bf16_t* src =
          qkv + (rowbase + (size_t)j * 32 + kr) * QKV_N + kcol + dc;
      *(uint4*)&Ksh[kr * KPAD + dc] = *(const uint4*)src;
    }
    for (int c = tid; c < 512; c += 256) {
      int kr = c >> 4;
      int dc = (c & 15) * 8;
      const bf16_t* src =
          qkv + (rowbase + (size_t)j * 32 + kr) * QKV_N + vcol + dc;
      uint4 raw = *(const uint4*)src;
      const bf16_t* e = (const bf16_t*)&raw;
#pragma unroll
      for (int x = 0; x < 8; ++x) Vsh[(dc + x) * VPAD + kr] = e[x];  // transpose
    }
    __syncthreads();

    // scores S = Q(16x128) . Kblk^T -> 16x32 (2 WMMA n-tiles, 4 k-chunks)
    v8f sc[2] = {};
#pragma unroll
    for (int t = 0; t < 2; ++t)
#pragma unroll
      for (int c = 0; c < 4; ++c) {
        v16bf bk = load_b_frag(Ksh, KPAD, t * 16, c * 32, lane);
        sc[t] = wmma_bf16(aq[c], bk, sc[t]);
      }

    // causal mask + online softmax (row = v + 8*h across lanes 0-15 / 16-31)
    float p0s[8], p1s[8];
#pragma unroll
    for (int v = 0; v < 8; ++v) {
      int q = qbase + v + 8 * h;
      int key0 = j * 32 + nn;
      float s0 = (key0      <= q) ? sc[0][v] : -1e30f;
      float s1 = (key0 + 16 <= q) ? sc[1][v] : -1e30f;
      float rm = fmaxf(s0, s1);
#pragma unroll
      for (int off = 8; off >= 1; off >>= 1)
        rm = fmaxf(rm, __shfl_xor(rm, off, 32));
      float mn = fmaxf(m_run[v], rm);
      float alpha = __expf(m_run[v] - mn);
      float p0 = __expf(s0 - mn);
      float p1 = __expf(s1 - mn);
      float rsum = p0 + p1;
#pragma unroll
      for (int off = 8; off >= 1; off >>= 1)
        rsum += __shfl_xor(rsum, off, 32);
      l_run[v] = l_run[v] * alpha + rsum;
      m_run[v] = mn;
#pragma unroll
      for (int t = 0; t < 8; ++t) acc[t][v] *= alpha;
      p0s[v] = p0; p1s[v] = p1;
    }

    // Re-layout P (C layout -> A layout) through per-wave LDS scratch.
    bf16_t* pw = Psh[wave];
#pragma unroll
    for (int v = 0; v < 8; ++v) {
      pw[(v + 8 * h) * PPAD + nn]      = (bf16_t)p0s[v];
      pw[(v + 8 * h) * PPAD + 16 + nn] = (bf16_t)p1s[v];
    }
    // same-wave DS ordering: compiler inserts s_wait_dscnt before reads
    v16bf pa = load_a_frag(pw, PPAD, 0, 0, lane);

    // O += P(16x32) . Vblk(32x128): 8 WMMA d-tiles, B from transposed Vsh.
#pragma unroll
    for (int t = 0; t < 8; ++t) {
      v16bf bv = load_b_frag(Vsh, VPAD, t * 16, 0, lane);
      acc[t] = wmma_bf16(pa, bv, acc[t]);
    }
  }

  // epilogue: normalize and store bf16 context at column (kv*G+g)*128 + d
#pragma unroll
  for (int t = 0; t < 8; ++t)
#pragma unroll
    for (int v = 0; v < 8; ++v) {
      int q = qbase + v + 8 * h;
      float val = acc[t][v] / l_run[v];
      ctx[(rowbase + q) * EMB + qcol + t * 16 + nn] = (bf16_t)val;
    }
}

// ---------------------------------------------------------------------------
extern "C" void kernel_launch(void* const* d_in, const int* in_sizes, int n_in,
                              void* d_out, int out_size, void* d_ws, size_t ws_size,
                              hipStream_t stream) {
  const float* q       = (const float*)d_in[0];  // [2,2048,4096]
  const float* w_qkv   = (const float*)d_in[1];  // [6144,4096]
  const float* w_dense = (const float*)d_in[2];  // [4096,4096]
  float* out = (float*)d_out;                    // [2,2048,4096]

  // bf16 workspace layout (~196 MB)
  bf16_t* qb    = (bf16_t*)d_ws;                            // 4096*4096
  bf16_t* wqkvb = qb    + (size_t)M_ROWS * EMB;             // 6144*4096
  bf16_t* wdb   = wqkvb + (size_t)QKV_N * EMB;              // 4096*4096
  bf16_t* qkvb  = wdb   + (size_t)EMB * EMB;                // 4096*6144
  bf16_t* ctxb  = qkvb  + (size_t)M_ROWS * QKV_N;           // 4096*4096

  // fp32 -> bf16 conversions
  {
    int n4;
    n4 = (M_ROWS * EMB) / 4;
    cvt_f32_bf16<<<(n4 + 255) / 256, 256, 0, stream>>>((const float4*)q, qb, n4);
    n4 = (QKV_N * EMB) / 4;
    cvt_f32_bf16<<<(n4 + 255) / 256, 256, 0, stream>>>((const float4*)w_qkv, wqkvb, n4);
    n4 = (EMB * EMB) / 4;
    cvt_f32_bf16<<<(n4 + 255) / 256, 256, 0, stream>>>((const float4*)w_dense, wdb, n4);
  }

  // QKV projection: [4096,4096] @ [6144,4096]^T -> bf16 [4096,6144]
  // (softmax scale folded into the Q columns, n < 4096)
  gemm_xwt<true><<<dim3(M_ROWS / 128, QKV_N / 128), 256, 0, stream>>>(
      qb, wqkvb, qkvb, M_ROWS, QKV_N, EMB, EMB, SOFTMAX_SCALE);

  // causal GQA flash attention -> bf16 context [4096,4096]
  attn_kernel<<<B_DIM * KVHEADS * (S_DIM / 32), 256, 0, stream>>>(qkvb, ctxb);

  // dense projection: [4096,4096] @ [4096,4096]^T -> fp32 out
  gemm_xwt<false><<<dim3(M_ROWS / 128, EMB / 128), 256, 0, stream>>>(
      ctxb, wdb, out, M_ROWS, EMB, EMB, 0, 1.0f);
}
